// StandaloneDetrTransformerDecoderLayer_2860448219657
// MI455X (gfx1250) — compile-verified, matrix operand order checked by reading, not run on hardware
//
#include <hip/hip_runtime.h>
#include <hip/hip_bf16.h>

// ---------------------------------------------------------------------------
// Problem constants (from the reference)
// ---------------------------------------------------------------------------
#define K_C      256
#define K_NH     8
#define K_HD     32
#define K_P      4
#define K_FFN    512
#define K_HBEV   200
#define K_WBEV   100
#define K_NQ     1000
#define K_BS     4
#define K_NV     (K_HBEV * K_WBEV)      // 20000
#define K_NQP    1024                   // padded queries per batch (pow2, /32)
#define K_M1     (K_BS * K_NQP)         // 4096 padded query rows
#define K_MV     (K_BS * K_NV)          // 80000 value rows
#define ATT_SCALE 0.17677669529663687f  // 1/sqrt(32)

typedef __attribute__((ext_vector_type(16))) _Float16 v16h;
typedef __attribute__((ext_vector_type(8)))  _Float16 v8h;
typedef __attribute__((ext_vector_type(8)))  float    v8f;

__device__ __forceinline__ v8f v8f_zero() {
  v8f r;
#pragma unroll
  for (int i = 0; i < 8; ++i) r[i] = 0.f;
  return r;
}

// Load one 16x32 f16 A/B fragment half-pair: per-lane 8 f16 at k0..k0+7 and
// k0+16..k0+23 (lane>=16 handles the +8 K-half).  Two global_load_b128.
__device__ __forceinline__ v16h load_frag16(const _Float16* __restrict__ p0) {
  const v8h lo = *reinterpret_cast<const v8h*>(p0);
  const v8h hi = *reinterpret_cast<const v8h*>(p0 + 16);
  v16h r;
#pragma unroll
  for (int i = 0; i < 8; ++i) { r[i] = lo[i]; r[i + 8] = hi[i]; }
  return r;
}
__device__ __forceinline__ const _Float16* frag_ptr(const _Float16* __restrict__ base,
                                                    int ld, int row0, int k0, int lane) {
  return base + (size_t)(row0 + (lane & 15)) * ld + k0 + ((lane >> 4) << 3);
}

// ---------------------------------------------------------------------------
// Generic WMMA GEMM:  out[M,N] = A[M,K] (f16) * W[N,K]^T (f16) + bias
// One wave computes a (16*MTR) x (16*NTR) tile.  Epilogues:
//   EPI 0: f32 store        EPI 1: f16 store
//   EPI 2: f16 ReLU store   EPI 3: f16 transposed store -> Vt[b][h][d][q]
// ---------------------------------------------------------------------------
template <int EPI, int NTR, int MTR>
__global__ void __launch_bounds__(256)
gemm_wmma(const _Float16* __restrict__ A, const _Float16* __restrict__ W,
          const float* __restrict__ bias, void* __restrict__ outp,
          int M, int N, int K) {
  const int lane = threadIdx.x & 31;
  const int wave = threadIdx.x >> 5;
  const int job  = blockIdx.x * 8 + wave;
  const int jobsPerM = N / (16 * NTR);
  const int njobs    = (M / (16 * MTR)) * jobsPerM;
  if (job >= njobs) return;                    // wave-uniform
  const int mt = (job / jobsPerM) * MTR;       // row tile index (16-row units)
  const int n0 = (job % jobsPerM) * (16 * NTR);

  v8f acc[MTR][NTR];
#pragma unroll
  for (int mm = 0; mm < MTR; ++mm)
#pragma unroll
    for (int t = 0; t < NTR; ++t) acc[mm][t] = v8f_zero();

  for (int k0 = 0; k0 < K; k0 += 32) {
    v16h a[MTR];
#pragma unroll
    for (int mm = 0; mm < MTR; ++mm)
      a[mm] = load_frag16(frag_ptr(A, K, (mt + mm) * 16, k0, lane));
    if (k0 + 32 < K) {
      // hide HBM latency on the streaming A operand (global_prefetch_b8)
#pragma unroll
      for (int mm = 0; mm < MTR; ++mm)
        __builtin_prefetch((const void*)frag_ptr(A, K, (mt + mm) * 16, k0 + 32, lane), 0, 1);
    }
#pragma unroll
    for (int t = 0; t < NTR; ++t) {
      const v16h b = load_frag16(frag_ptr(W, K, n0 + t * 16, k0, lane));
#pragma unroll
      for (int mm = 0; mm < MTR; ++mm)
        acc[mm][t] = __builtin_amdgcn_wmma_f32_16x16x32_f16(
            false, a[mm], false, b, (short)0, acc[mm][t], false, false);
    }
  }

#pragma unroll
  for (int mm = 0; mm < MTR; ++mm) {
#pragma unroll
    for (int t = 0; t < NTR; ++t) {
      const int n  = n0 + t * 16 + (lane & 15);
      const float bv = bias ? bias[n] : 0.f;
      const int mb = (mt + mm) * 16 + ((lane >> 4) << 3);
#pragma unroll
      for (int r = 0; r < 8; ++r) {
        float v = acc[mm][t][r] + bv;
        const int m = mb + r;
        if (EPI == 0) {
          ((float*)outp)[(size_t)m * N + n] = v;
        } else if (EPI == 1) {
          ((_Float16*)outp)[(size_t)m * N + n] = (_Float16)v;
        } else if (EPI == 2) {
          ((_Float16*)outp)[(size_t)m * N + n] = (_Float16)fmaxf(v, 0.f);
        } else {  // Vt[b][h][d][q], NQP columns
          const int b = m >> 10, q = m & (K_NQP - 1);
          const int h = n >> 5,  d = n & 31;
          ((_Float16*)outp)[((((size_t)b * K_NH + h) * K_HD + d) << 10) + q] = (_Float16)v;
        }
      }
    }
  }
}

// ---------------------------------------------------------------------------
// Flash attention: wave per (b, h, 16-query tile).  HD=32 -> 1 WMMA per score
// tile; online softmax; P re-layout via per-wave LDS; 2 WMMAs for P*V.
// ---------------------------------------------------------------------------
__global__ void __launch_bounds__(256)
flash_attn(const _Float16* __restrict__ qh, const _Float16* __restrict__ kh,
           const _Float16* __restrict__ vt, _Float16* __restrict__ attn) {
  __shared__ __align__(16) _Float16 pbuf[8][16][32];
  const int lane = threadIdx.x & 31;
  const int wave = threadIdx.x >> 5;
  const int job  = blockIdx.x * 8 + wave;
  const int nqt  = K_NQP / 16;                       // 64
  if (job >= K_BS * K_NH * nqt) return;              // wave-uniform
  const int qt = job % nqt;
  const int h  = (job / nqt) % K_NH;
  const int b  = job / (nqt * K_NH);
  const int rowbase = b * K_NQP;

  const v16h qf = load_frag16(frag_ptr(qh + h * K_HD, K_C, rowbase + qt * 16, 0, lane));
  const _Float16* vbase = vt + (((size_t)b * K_NH + h) * K_HD << 10);

  v8f o0 = v8f_zero(), o1 = v8f_zero();
  float rowm[8], rowl[8];
#pragma unroll
  for (int r = 0; r < 8; ++r) { rowm[r] = -3.0e38f; rowl[r] = 0.f; }

  for (int kc = 0; kc < K_NQP / 32; ++kc) {
    const v16h k0f = load_frag16(frag_ptr(kh + h * K_HD, K_C, rowbase + kc * 32, 0, lane));
    const v16h k1f = load_frag16(frag_ptr(kh + h * K_HD, K_C, rowbase + kc * 32 + 16, 0, lane));
    v8f s0 = __builtin_amdgcn_wmma_f32_16x16x32_f16(false, qf, false, k0f,
                                                    (short)0, v8f_zero(), false, false);
    v8f s1 = __builtin_amdgcn_wmma_f32_16x16x32_f16(false, qf, false, k1f,
                                                    (short)0, v8f_zero(), false, false);
    const int key0 = kc * 32 + (lane & 15);
    const int key1 = key0 + 16;
    float p0[8], p1[8];
#pragma unroll
    for (int r = 0; r < 8; ++r) {
      p0[r] = (key0 < K_NQ) ? s0[r] * ATT_SCALE : -1.0e30f;
      p1[r] = (key1 < K_NQ) ? s1[r] * ATT_SCALE : -1.0e30f;
    }
    // row max across the 16-lane group, then online-softmax update
#pragma unroll
    for (int r = 0; r < 8; ++r) {
      float tm = fmaxf(p0[r], p1[r]);
#pragma unroll
      for (int msk = 1; msk < 16; msk <<= 1) tm = fmaxf(tm, __shfl_xor(tm, msk, 32));
      const float mn = fmaxf(rowm[r], tm);
      const float alpha = __expf(rowm[r] - mn);
      rowm[r] = mn;
      p0[r] = __expf(p0[r] - mn);
      p1[r] = __expf(p1[r] - mn);
      float ts = p0[r] + p1[r];
#pragma unroll
      for (int msk = 1; msk < 16; msk <<= 1) ts += __shfl_xor(ts, msk, 32);
      rowl[r] = rowl[r] * alpha + ts;
      o0[r] *= alpha; o1[r] *= alpha;
    }
    // C-layout -> A-layout through LDS (per-wave region, DS ops in-order)
#pragma unroll
    for (int r = 0; r < 8; ++r) {
      const int m = r + ((lane >> 4) << 3);
      pbuf[wave][m][lane & 15]        = (_Float16)p0[r];
      pbuf[wave][m][16 + (lane & 15)] = (_Float16)p1[r];
    }
    const v16h pf  = load_frag16(&pbuf[wave][lane & 15][(lane >> 4) << 3]);
    const v16h vf0 = load_frag16(frag_ptr(vbase, K_NQP, 0,  kc * 32, lane));
    const v16h vf1 = load_frag16(frag_ptr(vbase, K_NQP, 16, kc * 32, lane));
    o0 = __builtin_amdgcn_wmma_f32_16x16x32_f16(false, pf, false, vf0, (short)0, o0, false, false);
    o1 = __builtin_amdgcn_wmma_f32_16x16x32_f16(false, pf, false, vf1, (short)0, o1, false, false);
  }
#pragma unroll
  for (int r = 0; r < 8; ++r) {
    const float inv = 1.0f / rowl[r];
    const int m = rowbase + qt * 16 + r + ((lane >> 4) << 3);
    attn[(size_t)m * K_C + h * K_HD + (lane & 15)]      = (_Float16)(o0[r] * inv);
    attn[(size_t)m * K_C + h * K_HD + 16 + (lane & 15)] = (_Float16)(o1[r] * inv);
  }
}

// ---------------------------------------------------------------------------
// Deformable sampling: wave per (b,q,h); lane = HD channel.
// ---------------------------------------------------------------------------
__global__ void __launch_bounds__(256)
msdeform_sample(const float* __restrict__ vproj, const float* __restrict__ so,
                const float* __restrict__ awr, const float* __restrict__ ref,
                _Float16* __restrict__ dout) {
  const int lane = threadIdx.x & 31;
  const int job  = blockIdx.x * 8 + (threadIdx.x >> 5);
  if (job >= K_BS * K_NQ * K_NH) return;
  const int h = job & 7;
  const int q = (job >> 3) % K_NQ;
  const int b = job / (K_NQ * K_NH);
  const int m = b * K_NQP + q;

  float a[K_P];
#pragma unroll
  for (int p = 0; p < K_P; ++p) a[p] = awr[(size_t)m * 32 + h * K_P + p];
  float mx = fmaxf(fmaxf(a[0], a[1]), fmaxf(a[2], a[3]));
  float sum = 0.f;
#pragma unroll
  for (int p = 0; p < K_P; ++p) { a[p] = __expf(a[p] - mx); sum += a[p]; }
  const float isum = 1.0f / sum;

  const float rx = ref[((size_t)b * K_NQ + q) * 2 + 0];
  const float ry = ref[((size_t)b * K_NQ + q) * 2 + 1];
  const float* vb = vproj + ((size_t)b * K_NV) * K_C + h * K_HD + lane;

  float acc = 0.f;
#pragma unroll
  for (int p = 0; p < K_P; ++p) {
    const float ox = so[(size_t)m * 64 + (h * K_P + p) * 2 + 0];
    const float oy = so[(size_t)m * 64 + (h * K_P + p) * 2 + 1];
    const float xi = (rx + ox * (1.0f / K_WBEV)) * K_WBEV - 0.5f;
    const float yi = (ry + oy * (1.0f / K_HBEV)) * K_HBEV - 0.5f;
    const float x0f = floorf(xi), y0f = floorf(yi);
    const float fx = xi - x0f, fy = yi - y0f;
    const int x0 = (int)x0f, y0 = (int)y0f;
    float s = 0.f;
#pragma unroll
    for (int dy = 0; dy < 2; ++dy)
#pragma unroll
      for (int dx = 0; dx < 2; ++dx) {
        const int xx = x0 + dx, yy = y0 + dy;
        const float wgt = (dx ? fx : 1.f - fx) * (dy ? fy : 1.f - fy);
        const bool ok = (xx >= 0) && (xx < K_WBEV) && (yy >= 0) && (yy < K_HBEV);
        const int xc = min(max(xx, 0), K_WBEV - 1);
        const int yc = min(max(yy, 0), K_HBEV - 1);
        const float g = vb[(size_t)(yc * K_WBEV + xc) * K_C];
        s += ok ? g * wgt : 0.f;
      }
    acc += s * a[p] * isum;
  }
  dout[(size_t)m * K_C + h * K_HD + lane] = (_Float16)acc;
}

// ---------------------------------------------------------------------------
// Residual add + LayerNorm (wave per row).  MODE 1: also emit f16(x+qpos)
// MODE 2: also emit f16(x)   MODE 3: final transposed f32 store to d_out.
// ---------------------------------------------------------------------------
template <int MODE>
__global__ void __launch_bounds__(256)
add_ln(const float* __restrict__ xin, const float* __restrict__ resid,
       const float* __restrict__ g, const float* __restrict__ beta,
       const float* __restrict__ qpos, float* __restrict__ xout,
       _Float16* __restrict__ hout, float* __restrict__ fout) {
  const int lane = threadIdx.x & 31;
  const int row  = blockIdx.x * 8 + (threadIdx.x >> 5);
  if (row >= K_M1) return;
  float v[8];
  const float* px = xin + (size_t)row * K_C + lane * 8;
  const float* pr = resid + (size_t)row * K_C + lane * 8;
  float s = 0.f, s2 = 0.f;
#pragma unroll
  for (int j = 0; j < 8; ++j) { v[j] = px[j] + pr[j]; s += v[j]; s2 += v[j] * v[j]; }
#pragma unroll
  for (int msk = 1; msk < 32; msk <<= 1) { s += __shfl_xor(s, msk, 32); s2 += __shfl_xor(s2, msk, 32); }
  const float mean = s * (1.0f / K_C);
  const float var  = s2 * (1.0f / K_C) - mean * mean;
  const float rstd = rsqrtf(var + 1e-5f);
  const int b = row >> 10, q = row & (K_NQP - 1);
#pragma unroll
  for (int j = 0; j < 8; ++j) {
    const int c = lane * 8 + j;
    const float y = (v[j] - mean) * rstd * g[c] + beta[c];
    if (MODE != 3) xout[(size_t)row * K_C + c] = y;
    if (MODE == 1) {
      const float qp = (q < K_NQ) ? qpos[((size_t)q * K_BS + b) * K_C + c] : 0.f;
      hout[(size_t)row * K_C + c] = (_Float16)(y + qp);
    } else if (MODE == 2) {
      hout[(size_t)row * K_C + c] = (_Float16)y;
    } else if (q < K_NQ) {
      fout[((size_t)q * K_BS + b) * K_C + c] = y;
    }
  }
}

// ---------------------------------------------------------------------------
// Packing / conversion kernels
// ---------------------------------------------------------------------------
__global__ void __launch_bounds__(256)
pack_query(const float* __restrict__ query, const float* __restrict__ qpos,
           float* __restrict__ xbuf, _Float16* __restrict__ q1,
           _Float16* __restrict__ x1) {
  const int i = blockIdx.x * 256 + threadIdx.x;
  if (i >= K_M1 * K_C) return;
  const int row = i >> 8, c = i & 255;
  const int b = row >> 10, q = row & (K_NQP - 1);
  float xv = 0.f, pv = 0.f;
  if (q < K_NQ) {
    const size_t src = ((size_t)q * K_BS + b) * K_C + c;
    xv = query[src]; pv = qpos[src];
  }
  xbuf[i] = xv;
  q1[i] = (_Float16)(xv + pv);
  x1[i] = (_Float16)xv;
}

__global__ void __launch_bounds__(256)
pack_value(const float* __restrict__ value, _Float16* __restrict__ vh) {
  const int i = blockIdx.x * 256 + threadIdx.x;
  if (i >= K_MV * K_C) return;
  const int row = i >> 8, c = i & 255;
  const int b = row / K_NV, n = row % K_NV;
  vh[i] = (_Float16)value[((size_t)n * K_BS + b) * K_C + c];
}

__global__ void __launch_bounds__(256)
f32_to_f16(const float* __restrict__ in, _Float16* __restrict__ out, int n) {
  const int i = blockIdx.x * 256 + threadIdx.x;
  if (i < n) out[i] = (_Float16)in[i];
}

// ---------------------------------------------------------------------------
// Host launcher
// ---------------------------------------------------------------------------
extern "C" void kernel_launch(void* const* d_in, const int* in_sizes, int n_in,
                              void* d_out, int out_size, void* d_ws, size_t ws_size,
                              hipStream_t stream) {
  (void)in_sizes; (void)n_in; (void)out_size; (void)ws_size;
  const float* query   = (const float*)d_in[0];
  const float* value   = (const float*)d_in[1];
  const float* qpos    = (const float*)d_in[2];
  const float* refpts  = (const float*)d_in[3];
  const float* inprj_w = (const float*)d_in[6];
  const float* inprj_b = (const float*)d_in[7];
  const float* mha_ow  = (const float*)d_in[8];
  const float* mha_ob  = (const float*)d_in[9];
  const float* so_w    = (const float*)d_in[10];
  const float* so_b    = (const float*)d_in[11];
  const float* aw_w    = (const float*)d_in[12];
  const float* aw_b    = (const float*)d_in[13];
  const float* vp_w    = (const float*)d_in[14];
  const float* vp_b    = (const float*)d_in[15];
  const float* op_w    = (const float*)d_in[16];
  const float* op_b    = (const float*)d_in[17];
  const float* ffn_w1  = (const float*)d_in[18];
  const float* ffn_b1  = (const float*)d_in[19];
  const float* ffn_w2  = (const float*)d_in[20];
  const float* ffn_b2  = (const float*)d_in[21];
  const float* ln1_g   = (const float*)d_in[22];
  const float* ln1_b   = (const float*)d_in[23];
  const float* ln2_g   = (const float*)d_in[24];
  const float* ln2_b   = (const float*)d_in[25];
  const float* ln3_g   = (const float*)d_in[26];
  const float* ln3_b   = (const float*)d_in[27];
  float* out = (float*)d_out;

  unsigned char* ws = (unsigned char*)d_ws;
  size_t off = 0;
  auto alloc = [&](size_t bytes) -> void* {
    void* p = ws + off;
    off += (bytes + 255) & ~(size_t)255;
    return p;
  };

  float*    xbuf  = (float*)alloc((size_t)K_M1 * K_C * 4);
  float*    mhaO  = (float*)alloc((size_t)K_M1 * K_C * 4);
  float*    dprO  = (float*)alloc((size_t)K_M1 * K_C * 4);
  float*    ffnO  = (float*)alloc((size_t)K_M1 * K_C * 4);
  float*    soBuf = (float*)alloc((size_t)K_M1 * 64 * 4);
  float*    awBuf = (float*)alloc((size_t)K_M1 * 32 * 4);
  float*    vproj = (float*)alloc((size_t)K_MV * K_C * 4);
  _Float16* q1h   = (_Float16*)alloc((size_t)K_M1 * K_C * 2);
  _Float16* x1h   = (_Float16*)alloc((size_t)K_M1 * K_C * 2);
  _Float16* q2h   = (_Float16*)alloc((size_t)K_M1 * K_C * 2);
  _Float16* x3h   = (_Float16*)alloc((size_t)K_M1 * K_C * 2);
  _Float16* qhh   = (_Float16*)alloc((size_t)K_M1 * K_C * 2);
  _Float16* khh   = (_Float16*)alloc((size_t)K_M1 * K_C * 2);
  _Float16* vth   = (_Float16*)alloc((size_t)K_BS * K_NH * K_HD * K_NQP * 2);
  _Float16* atth  = (_Float16*)alloc((size_t)K_M1 * K_C * 2);
  _Float16* defh  = (_Float16*)alloc((size_t)K_M1 * K_C * 2);
  _Float16* f1h   = (_Float16*)alloc((size_t)K_M1 * K_FFN * 2);
  _Float16* valh  = (_Float16*)alloc((size_t)K_MV * K_C * 2);
  _Float16* wqkvh = (_Float16*)alloc((size_t)768 * 256 * 2);
  _Float16* mowh  = (_Float16*)alloc((size_t)256 * 256 * 2);
  _Float16* sowh  = (_Float16*)alloc((size_t)64 * 256 * 2);
  _Float16* awwh  = (_Float16*)alloc((size_t)32 * 256 * 2);
  _Float16* vpwh  = (_Float16*)alloc((size_t)256 * 256 * 2);
  _Float16* opwh  = (_Float16*)alloc((size_t)256 * 256 * 2);
  _Float16* w1h   = (_Float16*)alloc((size_t)512 * 256 * 2);
  _Float16* w2h   = (_Float16*)alloc((size_t)256 * 512 * 2);

  auto cvt = [&](const float* src, _Float16* dst, int n) {
    f32_to_f16<<<(n + 255) / 256, 256, 0, stream>>>(src, dst, n);
  };
  cvt(inprj_w, wqkvh, 768 * 256);
  cvt(mha_ow,  mowh,  256 * 256);
  cvt(so_w,    sowh,  64 * 256);
  cvt(aw_w,    awwh,  32 * 256);
  cvt(vp_w,    vpwh,  256 * 256);
  cvt(op_w,    opwh,  256 * 256);
  cvt(ffn_w1,  w1h,   512 * 256);
  cvt(ffn_w2,  w2h,   256 * 512);

  pack_query<<<(K_M1 * K_C + 255) / 256, 256, 0, stream>>>(query, qpos, xbuf, q1h, x1h);
  pack_value<<<(K_MV * K_C + 255) / 256, 256, 0, stream>>>(value, valh);

#define LAUNCH_GEMM(EPI, NTR, MTR, A, W, B, O, M, N, K)                         \
  do {                                                                          \
    int _jobs = ((M) / (16 * (MTR))) * ((N) / (16 * (NTR)));                    \
    gemm_wmma<EPI, NTR, MTR><<<(_jobs + 7) / 8, 256, 0, stream>>>(A, W, B, O, M, N, K); \
  } while (0)

  // --- Stage 1: MHA ---
  LAUNCH_GEMM(1, 4, 2, q1h, wqkvh,             inprj_b,       qhh,  K_M1, 256, 256);
  LAUNCH_GEMM(1, 4, 2, q1h, wqkvh + 256 * 256, inprj_b + 256, khh,  K_M1, 256, 256);
  LAUNCH_GEMM(3, 4, 2, x1h, wqkvh + 512 * 256, inprj_b + 512, vth,  K_M1, 256, 256);
  {
    int jobs = K_BS * K_NH * (K_NQP / 16);
    flash_attn<<<(jobs + 7) / 8, 256, 0, stream>>>(qhh, khh, vth, atth);
  }
  LAUNCH_GEMM(0, 4, 2, atth, mowh, mha_ob, mhaO, K_M1, 256, 256);
  add_ln<1><<<K_M1 / 8, 256, 0, stream>>>(xbuf, mhaO, ln1_g, ln1_b, qpos, xbuf, q2h, nullptr);

  // --- Stage 2: MS-deformable attention ---
  LAUNCH_GEMM(0, 4, 2, valh, vpwh, vp_b, vproj, K_MV, 256, 256);
  LAUNCH_GEMM(0, 4, 2, q2h, sowh, so_b, soBuf, K_M1, 64, 256);
  LAUNCH_GEMM(0, 2, 2, q2h, awwh, aw_b, awBuf, K_M1, 32, 256);
  {
    int jobs = K_BS * K_NQ * K_NH;
    msdeform_sample<<<(jobs + 7) / 8, 256, 0, stream>>>(vproj, soBuf, awBuf, refpts, defh);
  }
  LAUNCH_GEMM(0, 4, 2, defh, opwh, op_b, dprO, K_M1, 256, 256);
  add_ln<2><<<K_M1 / 8, 256, 0, stream>>>(xbuf, dprO, ln2_g, ln2_b, nullptr, xbuf, x3h, nullptr);

  // --- Stage 3: FFN ---
  LAUNCH_GEMM(2, 4, 2, x3h, w1h, ffn_b1, f1h, K_M1, 512, 256);
  LAUNCH_GEMM(0, 4, 2, f1h, w2h, ffn_b2, ffnO, K_M1, 256, 512);
  add_ln<3><<<K_M1 / 8, 256, 0, stream>>>(xbuf, ffnO, ln3_g, ln3_b, nullptr, nullptr, nullptr, out);

#undef LAUNCH_GEMM
}
